// SelfSupervisedRGCN_23656679866470
// MI455X (gfx1250) — matrix operand order repaired and checked by reading, not compiled, not run on hardware
//
#include <hip/hip_runtime.h>

// ---------------------------------------------------------------------------
// RGCN forward for MI455X (gfx1250, wave32).
// Strategy:
//   * bf16 WMMA (v_wmma_f32_16x16x32_bf16) for all GEMMs, f32 accumulate.
//   * Per-relation pipelining: xw_r (51.2MB) + agg (51.2MB) stay L2-resident
//     (192MB L2), so edge gather + atomic scatter hit L2, not HBM.
//   * Root GEMM initializes agg (+bias); edges atomicAdd into it.
// ---------------------------------------------------------------------------

typedef __attribute__((ext_vector_type(16))) __bf16 v16bf;
typedef __attribute__((ext_vector_type(8)))  float  v8f;

#define NREL 8

// ----------------------------- small utility kernels -----------------------

__global__ void zero_f32_kernel(float* p, long n) {
  long i = (long)blockIdx.x * blockDim.x + threadIdx.x;
  if (i < n) p[i] = 0.0f;
}

__global__ void count_edges_kernel(const int* __restrict__ dst,
                                   const int* __restrict__ et,
                                   float* __restrict__ cnt, int E) {
  int e = blockIdx.x * blockDim.x + threadIdx.x;
  if (e < E) atomicAdd(&cnt[dst[e] * NREL + et[e]], 1.0f);
}

__global__ void inv_kernel(float* c, long n) {
  long i = (long)blockIdx.x * blockDim.x + threadIdx.x;
  if (i < n) c[i] = 1.0f / fmaxf(c[i], 1.0f);
}

// h_b[n, j] = bf16( emb[x[n]][j] )
__global__ void gather_emb_kernel(const int* __restrict__ x,
                                  const float* __restrict__ emb,
                                  __bf16* __restrict__ hb, int N, int D) {
  long i = (long)blockIdx.x * blockDim.x + threadIdx.x;
  if (i >= (long)N * D) return;
  long n = i / D, j = i % D;
  hb[i] = (__bf16)emb[(long)x[n] * D + j];
}

__global__ void relu_cvt_kernel(const float* __restrict__ in,
                                __bf16* __restrict__ out, long n) {
  long i = (long)blockIdx.x * blockDim.x + threadIdx.x;
  if (i < n) out[i] = (__bf16)fmaxf(in[i], 0.0f);
}

// Pack W[r][k][n] (f32 row-major) into WMMA B-fragment order:
//   Bp[r][nt][kt][lane][j] = bf16( W[r][kt*32 + 16*(lane/16) + j][nt*16 + lane%16] )
// so the GEMM can load a lane's 16 bf16 B values as two contiguous uint4.
__global__ void pack_b_kernel(const float* __restrict__ W,
                              __bf16* __restrict__ Bp,
                              int R, int K, int Nout) {
  long idx = (long)blockIdx.x * blockDim.x + threadIdx.x;
  long per_r = (long)K * Nout;
  if (idx >= (long)R * per_r) return;
  int KT = K >> 5;
  long r = idx / per_r;
  long t = idx % per_r;
  int j    = (int)(t & 15);
  int lane = (int)((t >> 4) & 31);
  int kt   = (int)((t >> 9) % KT);
  int nt   = (int)((t >> 9) / KT);
  int n = nt * 16 + (lane & 15);
  int k = kt * 32 + (lane >> 4) * 16 + j;
  Bp[idx] = (__bf16)W[r * per_r + (long)k * Nout + n];
}

// ----------------------------- WMMA GEMM -----------------------------------
// C[M, NT*16] = A[M, KT*32] (bf16) * Bpacked (+ bias).  One wave per 16-row
// M-tile; A fragments for the whole K cached in registers; loops over N-tiles.
template <int KT>
__global__ void wmma_gemm_kernel(const __bf16* __restrict__ A,
                                 const __bf16* __restrict__ Bp,
                                 const float* __restrict__ bias,
                                 float* __restrict__ C,
                                 int Mtiles, int NT) {
  const int K = KT * 32;
  int wave = blockIdx.x * (blockDim.x >> 5) + (threadIdx.x >> 5);
  if (wave >= Mtiles) return;
  int lane = threadIdx.x & 31;
  int mlo  = lane & 15;       // row within tile for A, column for C
  int hi   = lane >> 4;

  // A fragment loads: lane holds row (wave*16 + mlo), two 16B chunks per kt
  // at element offsets kt*32 + 16*hi and kt*32 + 16*hi + 8.
  const uint4* arow =
      reinterpret_cast<const uint4*>(A + (long)(wave * 16 + mlo) * K);
  v16bf afrag[KT];
#pragma unroll
  for (int kt = 0; kt < KT; ++kt) {
    union { v16bf v; uint4 u[2]; } t;
    t.u[0] = arow[kt * 4 + 2 * hi];
    t.u[1] = arow[kt * 4 + 2 * hi + 1];
    afrag[kt] = t.v;
  }

  const int ldc = NT * 16;
  const uint4* bbase = reinterpret_cast<const uint4*>(Bp);
  for (int nt = 0; nt < NT; ++nt) {
    v8f acc = {};
    const uint4* bp = bbase + ((long)(nt * KT) * 32 + lane) * 2;
#pragma unroll
    for (int kt = 0; kt < KT; ++kt) {
      union { v16bf v; uint4 u[2]; } b;
      b.u[0] = bp[0];
      b.u[1] = bp[1];
      bp += 64;  // 32 lanes * 2 uint4 per k-tile
      acc = __builtin_amdgcn_wmma_f32_16x16x32_bf16(
          false, afrag[kt], false, b.v, (short)0, acc, false, false);
    }
    if (bias) {
      float bv = bias[nt * 16 + mlo];
#pragma unroll
      for (int i = 0; i < 8; ++i) acc[i] += bv;
    }
    // C element (m = i + 8*hi, n = nt*16 + mlo)
    float* cp = C + (long)(wave * 16 + hi * 8) * ldc + nt * 16 + mlo;
#pragma unroll
    for (int i = 0; i < 8; ++i) cp[(long)i * ldc] = acc[i];
  }
}

// ----------------------------- edge scatter --------------------------------
// One wave per edge; lanes stride the feature dim.  agg[dst] += xw_r[src]*inv.
__global__ void scatter_kernel(const int* __restrict__ src,
                               const int* __restrict__ dst,
                               const int* __restrict__ et,
                               const float* __restrict__ inv,
                               const float* __restrict__ xw,
                               float* __restrict__ agg,
                               int E, int D, int r) {
  int e = blockIdx.x * (blockDim.x >> 5) + (threadIdx.x >> 5);
  if (e >= E) return;
  int t = et[e];
  if (t != r) return;
  int lane = threadIdx.x & 31;
  int s = src[e], d = dst[e];
  float w = inv[d * NREL + t];
  const float* m = xw + (long)s * D;
  float* o = agg + (long)d * D;
  for (int j = lane; j < D; j += 32) atomicAdd(o + j, m[j] * w);
}

// ----------------------------- launcher ------------------------------------

static inline long cdiv(long a, long b) { return (a + b - 1) / b; }

extern "C" void kernel_launch(void* const* d_in, const int* in_sizes, int n_in,
                              void* d_out, int out_size, void* d_ws, size_t ws_size,
                              hipStream_t stream) {
  const int* x       = (const int*)d_in[0];
  const int* eidx    = (const int*)d_in[1];
  const int* etype   = (const int*)d_in[2];
  const float* emb   = (const float*)d_in[3];
  const float* W1    = (const float*)d_in[4];
  const float* root1 = (const float*)d_in[5];
  const float* b1    = (const float*)d_in[6];
  const float* W2    = (const float*)d_in[7];
  const float* root2 = (const float*)d_in[8];
  const float* b2    = (const float*)d_in[9];
  float* out = (float*)d_out;

  const int N = in_sizes[0];           // 50000
  const int E = in_sizes[2];           // 800000
  const int DIN = 128, DHID = 256, DOUT = 128;
  const int MT = N / 16;               // 3125 M-tiles (N divisible by 16)
  const int* esrc = eidx;
  const int* edst = eidx + E;

  // ---- workspace carve-up (all offsets 512B aligned) ----
  char* ws = (char*)d_ws;
  size_t cur = 0;
  auto alloc = [&](size_t bytes) {
    size_t off = cur; cur += (bytes + 511) & ~(size_t)511; return off;
  };
  float* inv   = (float*)(ws + alloc((size_t)N * NREL * 4));   //   1.6 MB
  __bf16* hb   = (__bf16*)(ws + alloc((size_t)N * DHID * 2));  //  25.6 MB
  float* xw    = (float*)(ws + alloc((size_t)N * DHID * 4));   //  51.2 MB
  float* agg   = (float*)(ws + alloc((size_t)N * DHID * 4));   //  51.2 MB
  __bf16* bpk  = (__bf16*)(ws + alloc((size_t)589824 * 2));    //   1.2 MB
  __bf16* bp_w1 = bpk;                  // 8*128*256
  __bf16* bp_r1 = bpk + 262144;         // 128*256
  __bf16* bp_w2 = bpk + 294912;         // 8*256*128
  __bf16* bp_r2 = bpk + 557056;         // 256*128
  (void)ws_size; (void)n_in; (void)out_size;

  const int TPB = 256;

  // ---- normalization table: inv[dst*8+etype] = 1/max(count,1) ----
  zero_f32_kernel<<<cdiv((long)N * NREL, TPB), TPB, 0, stream>>>(inv, (long)N * NREL);
  count_edges_kernel<<<cdiv(E, TPB), TPB, 0, stream>>>(edst, etype, inv, E);
  inv_kernel<<<cdiv((long)N * NREL, TPB), TPB, 0, stream>>>(inv, (long)N * NREL);

  // ---- pack weights into WMMA B-fragment order (bf16) ----
  pack_b_kernel<<<cdiv((long)NREL * DIN * DHID, TPB), TPB, 0, stream>>>(W1, bp_w1, NREL, DIN, DHID);
  pack_b_kernel<<<cdiv((long)DIN * DHID, TPB), TPB, 0, stream>>>(root1, bp_r1, 1, DIN, DHID);
  pack_b_kernel<<<cdiv((long)NREL * DHID * DOUT, TPB), TPB, 0, stream>>>(W2, bp_w2, NREL, DHID, DOUT);
  pack_b_kernel<<<cdiv((long)DHID * DOUT, TPB), TPB, 0, stream>>>(root2, bp_r2, 1, DHID, DOUT);

  // ---- h0 = bf16(emb_table[x]) ----
  gather_emb_kernel<<<cdiv((long)N * DIN, TPB), TPB, 0, stream>>>(x, emb, hb, N, DIN);

  const int gemm_blocks = (int)cdiv(MT, TPB / 32);
  const int scat_blocks = (int)cdiv(E, TPB / 32);

  // ================= layer 1 (K=128 -> KT=4, out 256 -> NT=16) =============
  // agg = h0 @ root1 + b1
  wmma_gemm_kernel<4><<<gemm_blocks, TPB, 0, stream>>>(hb, bp_r1, b1, agg, MT, 16);
  for (int r = 0; r < NREL; ++r) {
    wmma_gemm_kernel<4><<<gemm_blocks, TPB, 0, stream>>>(
        hb, bp_w1 + (long)r * DIN * DHID, nullptr, xw, MT, 16);
    scatter_kernel<<<scat_blocks, TPB, 0, stream>>>(
        esrc, edst, etype, inv, xw, agg, E, DHID, r);
  }
  // h1 = bf16(relu(agg))
  relu_cvt_kernel<<<cdiv((long)N * DHID, TPB), TPB, 0, stream>>>(agg, hb, (long)N * DHID);

  // ================= layer 2 (K=256 -> KT=8, out 128 -> NT=8) ==============
  // out = h1 @ root2 + b2, then edges accumulate straight into d_out
  wmma_gemm_kernel<8><<<gemm_blocks, TPB, 0, stream>>>(hb, bp_r2, b2, out, MT, 8);
  for (int r = 0; r < NREL; ++r) {
    wmma_gemm_kernel<8><<<gemm_blocks, TPB, 0, stream>>>(
        hb, bp_w2 + (long)r * DHID * DOUT, nullptr, xw, MT, 8);
    scatter_kernel<<<scat_blocks, TPB, 0, stream>>>(
        esrc, edst, etype, inv, xw, out, E, DOUT, r);
  }
}
